// CompressedModel_21045339750947
// MI455X (gfx1250) — matrix operand order repaired and checked by reading, not compiled, not run on hardware
//
#include <hip/hip_runtime.h>
#include <math.h>

// ---------------------------------------------------------------------------
// Token-merge pipeline for MI455X (gfx1250, wave32, WMMA).
// Dominant cost: sim = xn @ xn^T (8 x 2048x2048x768 ~ 51.5 GFLOP). We fuse
// elu+row-mean into the GEMM so the 134MB sim matrix is never materialized;
// xn (bf16, 25MB) stays L2-resident, making this compute-bound on the bf16
// WMMA pipe. B tiles are streamed with GLOBAL_LOAD_ASYNC_TO_LDS_B128
// (ASYNCcnt) double-buffered against the WMMA compute.
// ---------------------------------------------------------------------------

typedef __attribute__((ext_vector_type(16))) __bf16 v16bf;
typedef __attribute__((ext_vector_type(8)))  float  v8f;
typedef int v4i __attribute__((vector_size(16)));

constexpr int   Bn = 8;
constexpr int   Tn = 2048;
constexpr int   Cn = 768;
constexpr int   Rn = 102;           // floor(T - T*0.95)
constexpr float MARGIN = 0.3f;
constexpr int   KT = Cn / 32;       // 24 k-tiles of K=32 per WMMA
constexpr int   ROWS_PER_WG = 128;  // 8 waves x 16 rows
constexpr int   NUM_CT = Tn / 16;   // 128 column tiles
constexpr int   LDS_A_ELTS = 8 * KT * 32 * 16;  // 98304 ushorts (192KB)
constexpr int   LDS_B_ELTS = KT * 32 * 16;      // 12288 ushorts (24KB)
constexpr int   LDS_BYTES  = (LDS_A_ELTS + 2 * LDS_B_ELTS) * 2;  // 245760 B
constexpr int   B_CHUNKS_PER_THREAD = (16 * (Cn / 8)) / 256;     // 6

#if __has_builtin(__builtin_amdgcn_global_load_async_to_lds_b128) && \
    __has_builtin(__builtin_amdgcn_s_wait_asynccnt)
#define USE_ASYNC 1
#else
#define USE_ASYNC 0
#endif

__device__ __forceinline__ void async_copy16(const void* g, void* l) {
#if USE_ASYNC
  __builtin_amdgcn_global_load_async_to_lds_b128(
      (__attribute__((address_space(1))) v4i*)g,
      (__attribute__((address_space(3))) v4i*)l, 0, 0);
#else
  *(uint4*)l = *(const uint4*)g;
#endif
}

template <int N>
__device__ __forceinline__ void wait_async() {
#if USE_ASYNC
  __builtin_amdgcn_s_wait_asynccnt(N);
#endif
}

__device__ __forceinline__ unsigned short f2bf_rne(float f) {
  unsigned int u = __float_as_uint(f);
  return (unsigned short)((u + 0x7FFFu + ((u >> 16) & 1u)) >> 16);
}
__device__ __forceinline__ float bf2f(unsigned short h) {
  return __uint_as_float(((unsigned int)h) << 16);
}

// --------------------------- 1) row-normalize ------------------------------
__global__ void k_normalize(const float* __restrict__ x,
                            unsigned short* __restrict__ xn) {
  const int row = blockIdx.x;                       // b*T + t
  const float* xr = x + (size_t)row * Cn;
  float ss = 0.f;
  for (int c = threadIdx.x; c < Cn; c += 256) { float v = xr[c]; ss += v * v; }
  __shared__ float red[256];
  red[threadIdx.x] = ss; __syncthreads();
  for (int s = 128; s > 0; s >>= 1) {
    if (threadIdx.x < s) red[threadIdx.x] += red[threadIdx.x + s];
    __syncthreads();
  }
  const float inv = rsqrtf(red[0]);
  unsigned short* xo = xn + (size_t)row * Cn;
  for (int c = threadIdx.x; c < Cn; c += 256) xo[c] = f2bf_rne(xr[c] * inv);
}

// ------------- 2) fused sim-GEMM + elu + row-mean (WMMA bf16) --------------
// Workgroup: 8 waves, 128 rows of xn vs all 2048 columns, K=768.
// A panel pre-swizzled to per-lane WMMA fragment order in LDS (loaded once,
// async); B tiles (16 cols x 768) double-buffered via async copies to LDS.
__global__ __launch_bounds__(256) void k_iso_wmma(
    const unsigned short* __restrict__ xn, const float* __restrict__ temp,
    float* __restrict__ iso) {
  extern __shared__ unsigned short smem[];
  unsigned short* As  = smem;                             // [wave][kt][lane][16]
  unsigned short* Bs0 = smem + LDS_A_ELTS;                // [kt][lane][16]
  unsigned short* Bs1 = smem + LDS_A_ELTS + LDS_B_ELTS;

  const int b = blockIdx.y;
  const int rowBase = blockIdx.x * ROWS_PER_WG;
  const int tid = threadIdx.x;
  const int wave = tid >> 5;
  const int lane = tid & 31;

  const float tc = fminf(fmaxf(temp[0], 0.001f), 0.05f);
  const float invT = 1.0f / tc;

  const unsigned short* xb = xn + (size_t)b * Tn * Cn;

  // A panel: 128 rows x 768. 16-bit A 16x32 layout: lane = half*16 + row;
  // lane elems e: 0..7 -> K = half*8 + e, 8..15 -> K = 16 + half*8 + (e-8).
  // Contiguous 8 bf16 (16B) in memory map to contiguous 8 elems in LDS.
  for (int c = tid; c < ROWS_PER_WG * (Cn / 8); c += 256) {   // 48 per thread
    const int lr = c / (Cn / 8);
    const int q  = c % (Cn / 8);
    const int k0 = q * 8;
    const int kt = k0 >> 5;
    const int kk = k0 & 31;
    const int half = (kk >> 3) & 1;
    const int e    = (kk >> 4) << 3;
    const int w  = lr >> 4;
    const int rt = lr & 15;
    const int ln = half * 16 + rt;
    async_copy16(xb + (size_t)(rowBase + lr) * Cn + k0,
                 As + ((((w * KT + kt) << 5) + ln) << 4) + e);
  }

  // B tile issue: 16 cols x 768 K. 16-bit B 32x16 layout: lane = half*16+col,
  // elem e: 0..7 -> K = half*8 + e, 8..15 -> K = 16 + half*8 + (e-8).
  auto issueB = [&](int ct, unsigned short* Bbuf) {
    for (int c = tid; c < 16 * (Cn / 8); c += 256) {          // 6 per thread
      const int col = c / (Cn / 8);
      const int q   = c % (Cn / 8);
      const int k0  = q * 8;
      const int kt  = k0 >> 5;
      const int kk  = k0 & 31;
      const int half = (kk >> 3) & 1;
      const int e    = (kk >> 4) << 3;  // 0 or 8
      const int ln   = half * 16 + col;
      async_copy16(xb + (size_t)(ct * 16 + col) * Cn + k0,
                   Bbuf + (((kt << 5) + ln) << 4) + e);
    }
  };

  issueB(0, Bs0);

  v8f rowsum = {};
  for (int ct = 0; ct < NUM_CT; ++ct) {
    unsigned short* cur = (ct & 1) ? Bs1 : Bs0;
    unsigned short* nxt = (ct & 1) ? Bs0 : Bs1;
    if (ct + 1 < NUM_CT) {
      issueB(ct + 1, nxt);
      wait_async<B_CHUNKS_PER_THREAD>();  // tile ct (and A panel) retired
    } else {
      wait_async<0>();
    }
    __syncthreads();  // tile-ct data from all waves visible

    v8f acc = {};
    const unsigned short* aPtr = As + ((((wave * KT) << 5) + lane) << 4);
    const unsigned short* bPtr = cur + (lane << 4);
    for (int kt = 0; kt < KT; ++kt) {
      v16bf a  = *(const v16bf*)(aPtr + (kt << 9));
      v16bf bm = *(const v16bf*)(bPtr + (kt << 9));
      acc = __builtin_amdgcn_wmma_f32_16x16x32_bf16(
          false, a, false, bm, (short)0, acc, false, false);
    }
    // fused elu((sim - margin)/temp) + row accumulation
#pragma unroll
    for (int g = 0; g < 8; ++g) {
      float z = (acc[g] - MARGIN) * invT;
      rowsum[g] += (z > 0.f) ? z : (__expf(z) - 1.0f);
    }
    __syncthreads();  // all waves done reading `cur` before it is refilled
  }

  // C-layout: lanes 0-15 hold (M=g, N=lane); lanes 16-31 hold (M=g+8).
  // Reduce over N within each 16-lane half.
#pragma unroll
  for (int g = 0; g < 8; ++g) {
    float v = rowsum[g];
    v += __shfl_xor(v, 1, 32);
    v += __shfl_xor(v, 2, 32);
    v += __shfl_xor(v, 4, 32);
    v += __shfl_xor(v, 8, 32);
    rowsum[g] = v;
  }
  const int mBase = rowBase + wave * 16 + ((lane >> 4) << 3);
  if ((lane & 15) == 0) {
#pragma unroll
    for (int g = 0; g < 8; ++g)
      iso[(size_t)b * Tn + mBase + g] = rowsum[g] * (1.0f / Tn);
  }
}

// ------------------- 3) size = 1 - softmax(iso) per batch ------------------
__global__ void k_size(const float* __restrict__ iso, float* __restrict__ sz) {
  const int b = blockIdx.x;
  const float* ir = iso + (size_t)b * Tn;
  __shared__ float red[256];
  float mx = -1e30f;
  for (int i = threadIdx.x; i < Tn; i += 256) mx = fmaxf(mx, ir[i]);
  red[threadIdx.x] = mx; __syncthreads();
  for (int s = 128; s > 0; s >>= 1) {
    if (threadIdx.x < s) red[threadIdx.x] = fmaxf(red[threadIdx.x], red[threadIdx.x + s]);
    __syncthreads();
  }
  mx = red[0]; __syncthreads();
  float sm = 0.f;
  for (int i = threadIdx.x; i < Tn; i += 256) sm += __expf(ir[i] - mx);
  red[threadIdx.x] = sm; __syncthreads();
  for (int s = 128; s > 0; s >>= 1) {
    if (threadIdx.x < s) red[threadIdx.x] += red[threadIdx.x + s];
    __syncthreads();
  }
  const float invS = 1.0f / red[0];
  for (int i = threadIdx.x; i < Tn; i += 256)
    sz[(size_t)b * Tn + i] = 1.0f - __expf(ir[i] - mx) * invS;
}

// ------------- 4) stable descending argsort of iso (bitonic) ---------------
__global__ __launch_bounds__(1024) void k_sort(const float* __restrict__ iso,
                                               int* __restrict__ idx) {
  const int b = blockIdx.x;
  __shared__ float key[Tn];
  __shared__ int   val[Tn];
  for (int i = threadIdx.x; i < Tn; i += 1024) {
    key[i] = -iso[(size_t)b * Tn + i];  // ascending on -iso == descending iso
    val[i] = i;
  }
  __syncthreads();
  for (int k = 2; k <= Tn; k <<= 1) {
    for (int j = k >> 1; j > 0; j >>= 1) {
      for (int i = threadIdx.x; i < Tn; i += 1024) {
        const int ixj = i ^ j;
        if (ixj > i) {
          const bool up = ((i & k) == 0);
          float ka = key[i], kb = key[ixj];
          int   va = val[i], vb = val[ixj];
          const bool gt = (ka > kb) || (ka == kb && va > vb);  // stable ties
          if (gt == up) { key[i] = kb; key[ixj] = ka; val[i] = vb; val[ixj] = va; }
        }
      }
      __syncthreads();
    }
  }
  for (int i = threadIdx.x; i < Tn; i += 1024) idx[(size_t)b * Tn + i] = val[i];
}

// ---------- 5) dst_idx = argmax_j sim[a_i, b_j] (elu is monotonic) ---------
__global__ void k_scores(const unsigned short* __restrict__ xn,
                         const int* __restrict__ idx, int* __restrict__ dst) {
  const int b = blockIdx.y;
  const int i = blockIdx.x;  // 0..R-1
  const int tid = threadIdx.x;  // 128 threads
  const int ai = idx[(size_t)b * Tn + i];
  __shared__ float ss[128];
  __shared__ int   sj[128];
  float best = -1e30f;
  if (tid < Rn) {
    const int bj = idx[(size_t)b * Tn + Rn + tid];
    const unsigned short* ar = xn + ((size_t)b * Tn + ai) * Cn;
    const unsigned short* br = xn + ((size_t)b * Tn + bj) * Cn;
    float d = 0.f;
    for (int c = 0; c < Cn; ++c) d += bf2f(ar[c]) * bf2f(br[c]);
    best = d;
  }
  ss[tid] = best; sj[tid] = tid;
  __syncthreads();
  for (int s = 64; s > 0; s >>= 1) {
    if (tid < s) {
      if (ss[tid + s] > ss[tid] ||
          (ss[tid + s] == ss[tid] && sj[tid + s] < sj[tid])) {
        ss[tid] = ss[tid + s]; sj[tid] = sj[tid + s];
      }
    }
    __syncthreads();
  }
  if (tid == 0) dst[(size_t)b * Rn + i] = sj[0];
}

// --------------------------- 6) merge stages -------------------------------
__global__ void k_merge_init(const float* __restrict__ x,
                             const float* __restrict__ sz,
                             const int* __restrict__ idx,
                             float* __restrict__ num, float* __restrict__ den) {
  const int b = blockIdx.x / Rn, j = blockIdx.x % Rn;
  const int bj = idx[(size_t)b * Tn + Rn + j];
  const float s = sz[(size_t)b * Tn + bj];
  const float* xr = x + ((size_t)b * Tn + bj) * Cn;
  float* nr = num + ((size_t)b * Rn + j) * Cn;
  for (int c = threadIdx.x; c < Cn; c += 256) nr[c] = xr[c] * s;
  if (threadIdx.x == 0) den[(size_t)b * Rn + j] = s;
}

__global__ void k_merge_scatter(const float* __restrict__ x,
                                const float* __restrict__ sz,
                                const int* __restrict__ idx,
                                const int* __restrict__ dst,
                                float* __restrict__ num, float* __restrict__ den) {
  const int b = blockIdx.x / Rn, i = blockIdx.x % Rn;
  const int ai = idx[(size_t)b * Tn + i];
  const int j  = dst[(size_t)b * Rn + i];
  const float s = sz[(size_t)b * Tn + ai];
  const float* xr = x + ((size_t)b * Tn + ai) * Cn;
  float* nr = num + ((size_t)b * Rn + j) * Cn;
  for (int c = threadIdx.x; c < Cn; c += 256) atomicAdd(&nr[c], xr[c] * s);
  if (threadIdx.x == 0) atomicAdd(&den[(size_t)b * Rn + j], s);
}

__global__ void k_finalize(const float* __restrict__ x,
                           const int* __restrict__ idx,
                           const float* __restrict__ num,
                           const float* __restrict__ den,
                           float* __restrict__ out) {
  const int b = blockIdx.y;
  const int m = blockIdx.x;  // 0 .. T-R-1
  float* orow = out + ((size_t)b * (Tn - Rn) + m) * Cn;
  if (m < Tn - 2 * Rn) {  // protected tokens: num/den == x
    const int src = idx[(size_t)b * Tn + 2 * Rn + m];
    const float* xr = x + ((size_t)b * Tn + src) * Cn;
    for (int c = threadIdx.x; c < Cn; c += 256) orow[c] = xr[c];
  } else {
    const int j = m - (Tn - 2 * Rn);
    const float invd = 1.0f / den[(size_t)b * Rn + j];
    const float* nr = num + ((size_t)b * Rn + j) * Cn;
    for (int c = threadIdx.x; c < Cn; c += 256) orow[c] = nr[c] * invd;
  }
}

// ------------------------------- launch ------------------------------------
extern "C" void kernel_launch(void* const* d_in, const int* in_sizes, int n_in,
                              void* d_out, int out_size, void* d_ws, size_t ws_size,
                              hipStream_t stream) {
  (void)in_sizes; (void)n_in; (void)out_size; (void)ws_size;
  const float* x    = (const float*)d_in[0];
  const float* temp = (const float*)d_in[1];
  float* out = (float*)d_out;

  char* ws = (char*)d_ws;
  size_t off = 0;
  auto alloc = [&](size_t bytes) -> void* {
    off = (off + 255) & ~(size_t)255;
    void* p = ws + off;
    off += bytes;
    return p;
  };
  unsigned short* xn  = (unsigned short*)alloc((size_t)Bn * Tn * Cn * 2);
  float* iso = (float*)alloc((size_t)Bn * Tn * 4);
  float* sz  = (float*)alloc((size_t)Bn * Tn * 4);
  int*   idx = (int*)alloc((size_t)Bn * Tn * 4);
  int*   dst = (int*)alloc((size_t)Bn * Rn * 4);
  float* den = (float*)alloc((size_t)Bn * Rn * 4);
  float* num = (float*)alloc((size_t)Bn * Rn * Cn * 4);

  (void)hipFuncSetAttribute((const void*)k_iso_wmma,
                            hipFuncAttributeMaxDynamicSharedMemorySize, LDS_BYTES);

  k_normalize<<<Bn * Tn, 256, 0, stream>>>(x, xn);
  k_iso_wmma<<<dim3(Tn / ROWS_PER_WG, Bn), 256, LDS_BYTES, stream>>>(xn, temp, iso);
  k_size<<<Bn, 256, 0, stream>>>(iso, sz);
  k_sort<<<Bn, 1024, 0, stream>>>(iso, idx);
  k_scores<<<dim3(Rn, Bn), 128, 0, stream>>>(xn, idx, dst);
  k_merge_init<<<Bn * Rn, 256, 0, stream>>>(x, sz, idx, num, den);
  k_merge_scatter<<<Bn * Rn, 256, 0, stream>>>(x, sz, idx, dst, num, den);
  k_finalize<<<dim3(Tn - Rn, Bn), 256, 0, stream>>>(x, idx, num, den, out);
}
